// ETGPDTransformer_67860483277142
// MI455X (gfx1250) — matrix operand order repaired and checked by reading, not compiled, not run on hardware
//
#include <hip/hip_runtime.h>
#include <hip/hip_bf16.h>

// ---------------------------------------------------------------------------
// ETGPD transformer for MI455X (gfx1250): f16 WMMA GEMMs + TDM tile staging
// + fused sparse attention.
// ---------------------------------------------------------------------------

#define Bsz   16
#define Tsz   1024
#define DinC  64
#define Dm    512
#define DffC  2048
#define Hn    8
#define Dh    64
#define Lc    4
#define TOPK  32
#define Ntok  (Bsz * Tsz)          // 16384

typedef __attribute__((ext_vector_type(16))) _Float16 v16h;
typedef __attribute__((ext_vector_type(8)))  float    v8f;
typedef __attribute__((ext_vector_type(4)))  unsigned v4u;
typedef __attribute__((ext_vector_type(8)))  int      v8i;
typedef __attribute__((ext_vector_type(4)))  int      v4i;

#if __has_builtin(__builtin_amdgcn_tensor_load_to_lds)
#define HAVE_TDM 1
#else
#define HAVE_TDM 0
#endif

static __device__ __forceinline__ v8f wmma_f16(v16h a, v16h b, v8f c) {
  return __builtin_amdgcn_wmma_f32_16x16x32_f16(false, a, false, b, (short)0, c,
                                                false, false);
}

// ---------------------------------------------------------------- convert --
__global__ void k_cvt(const float* __restrict__ s, _Float16* __restrict__ d, int n) {
  for (int i = blockIdx.x * blockDim.x + threadIdx.x; i < n;
       i += gridDim.x * blockDim.x)
    d[i] = (_Float16)s[i];
}

// ---------------------------------------------------------- input project --
// h = x @ W_in + b_in + PE ; writes f32 master + f16 copy
__global__ void k_inproj(const float* __restrict__ x, const float* __restrict__ W,
                         const float* __restrict__ b, float* __restrict__ hf,
                         _Float16* __restrict__ hb) {
  int idx = blockIdx.x * 256 + threadIdx.x;      // Ntok*Dm threads
  int j = idx & (Dm - 1);
  int n = idx >> 9;                              // token
  int t = n & (Tsz - 1);
  const float* xr = x + (size_t)n * DinC;
  float acc = b[j];
#pragma unroll 8
  for (int kk = 0; kk < DinC; ++kk) acc += xr[kk] * W[kk * Dm + j];
  int i2 = j >> 1;
  float div = __expf((float)(2 * i2) * (-9.210340371976184f / (float)Dm));
  float ang = (float)t * div;
  acc += (j & 1) ? __cosf(ang) : __sinf(ang);
  hf[idx] = acc;
  hb[idx] = (_Float16)acc;
}

// ------------------------------------------------------------------ GEMM ---
// C[M x Nc] = A[M x K] (f16) @ W[K x Nc] (f16), f32 accum.
// A tile (128 x 32) staged into LDS by the Tensor Data Mover with LDS row
// padding configured to a 40-half stride (bank-conflict free, matches the
// WMMA fragment loader). W tile transposed into LDS by the 8 waves.
// EPI 0: outf = acc + bias      (f32)
// EPI 1: outh = acc (+bias)     (f16)
// EPI 2: outh = gelu(acc+bias)  (f16)
template <int EPI>
__global__ __launch_bounds__(256) void k_gemm(const _Float16* __restrict__ A,
                                              const _Float16* __restrict__ W,
                                              const float* __restrict__ bias,
                                              float* __restrict__ outf,
                                              _Float16* __restrict__ outh,
                                              int M, int K, int Nc) {
  __shared__ _Float16 As[128 * 40];   // [m][k], stride 40 (pad via TDM)
  __shared__ _Float16 Wt[128 * 40];   // transposed: [n][k]
  const int t = threadIdx.x;
  const int lane = t & 31, wid = t >> 5;
  const int wm = wid >> 1, wn = wid & 1;          // 4x2 wave grid
  const int hi = lane >> 4, l16 = lane & 15;
  const int rb = blockIdx.y * 128, nb = blockIdx.x * 128;

  v8f acc[2][4];
  const v8f vz = {0.f, 0.f, 0.f, 0.f, 0.f, 0.f, 0.f, 0.f};
#pragma unroll
  for (int i = 0; i < 2; ++i)
#pragma unroll
    for (int j = 0; j < 4; ++j) acc[i][j] = vz;

  for (int k0 = 0; k0 < K; k0 += 32) {
    __syncthreads();
#if HAVE_TDM
    // ---- A tile via Tensor Data Mover (one wave issues the DMA) ----
    if (wid == 0) {
      unsigned lds = (unsigned)(size_t)(void*)As;
      unsigned long long ga =
          (unsigned long long)(size_t)(A + (size_t)rb * K + k0);
      v4u g0;
      g0[0] = 1u;                                  // count=1, user mode
      g0[1] = lds;                                 // lds_addr
      g0[2] = (unsigned)ga;                        // global_addr[31:0]
      g0[3] = (unsigned)((ga >> 32) & 0x01FFFFFFu) | (2u << 30);  // type=2
      v8i g1;
      // data_size=1 (2B), pad_enable, pad_interval=3 (16 dwords = one row),
      // pad_amount=3 (4 dwords) -> 40-half LDS row stride
      g1[0] = (int)((1u << 16) | (1u << 20) | (3u << 22) | (3u << 25));
      g1[1] = (int)(((unsigned)K & 0xFFFFu) << 16);          // tensor_dim0 lo
      g1[2] = (int)(((unsigned)K >> 16) |
                    (((unsigned)M & 0xFFFFu) << 16));        // dim0 hi|dim1 lo
      g1[3] = (int)(((unsigned)M >> 16) | (32u << 16));      // dim1 hi|tile0
      g1[4] = (int)128u;                                     // tile_dim1
      g1[5] = (int)(unsigned)K;                              // dim0_stride lo
      g1[6] = 0;                                             // stride hi
      g1[7] = 0;
      v4i gz4 = {0, 0, 0, 0};
      v8i gz8 = {0, 0, 0, 0, 0, 0, 0, 0};
      __builtin_amdgcn_tensor_load_to_lds(g0, g1, gz4, gz4, gz8, 0);
      __builtin_amdgcn_s_wait_tensorcnt(0);
    }
#else
    // fallback: manual A staging (uint2 slots of 4 halves)
    for (int s = t; s < 1024; s += 256) {
      int m = s >> 3, kk = (s & 7) * 4;
      uint2 d = *(const uint2*)(A + (size_t)(rb + m) * K + k0 + kk);
      unsigned* dst = (unsigned*)(As + m * 40 + kk);
      dst[0] = d.x; dst[1] = d.y;
    }
#endif
    // ---- stage W tile 32x128, transposed into Wt[n][k] ----
    {
      int kq = t >> 3;              // 0..31
      int n0 = (t & 7) * 16;        // 0..112
      const uint2* wr = (const uint2*)(W + (size_t)(k0 + kq) * Nc + nb + n0);
      _Float16 tmp[16];
      *(uint2*)&tmp[0]  = wr[0];
      *(uint2*)&tmp[4]  = wr[1];
      *(uint2*)&tmp[8]  = wr[2];
      *(uint2*)&tmp[12] = wr[3];
#pragma unroll
      for (int i = 0; i < 16; ++i) Wt[(n0 + i) * 40 + kq] = tmp[i];
    }
    __syncthreads();
    if (k0 + 32 < K)
      __builtin_prefetch(W + (size_t)(k0 + 32 + (t >> 3)) * Nc + nb, 0, 1);

    v16h af[2], bf[4];
#pragma unroll
    for (int i = 0; i < 2; ++i) {
      const _Float16* ar = As + (wm * 32 + i * 16 + l16) * 40;
      unsigned* au = (unsigned*)&af[i];
#pragma unroll
      for (int v = 0; v < 8; ++v)
        au[v] = *(const unsigned*)(ar + 2 * v + (v >= 4 ? 8 : 0) + hi * 8);
    }
#pragma unroll
    for (int j = 0; j < 4; ++j) {
      const _Float16* wr = Wt + (wn * 64 + j * 16 + l16) * 40;
      unsigned* bu = (unsigned*)&bf[j];
#pragma unroll
      for (int v = 0; v < 8; ++v)
        bu[v] = *(const unsigned*)(wr + 2 * v + (v >= 4 ? 8 : 0) + hi * 8);
    }
#pragma unroll
    for (int i = 0; i < 2; ++i)
#pragma unroll
      for (int j = 0; j < 4; ++j) acc[i][j] = wmma_f16(af[i], bf[j], acc[i][j]);
  }

#pragma unroll
  for (int i = 0; i < 2; ++i)
#pragma unroll
    for (int j = 0; j < 4; ++j) {
      int n = nb + wn * 64 + j * 16 + l16;
      float bv = bias ? bias[n] : 0.f;
      int mbase = rb + wm * 32 + i * 16 + hi * 8;
#pragma unroll
      for (int r = 0; r < 8; ++r) {
        float vv = acc[i][j][r] + bv;
        size_t o = (size_t)(mbase + r) * Nc + n;
        if (EPI == 0) outf[o] = vv;
        else if (EPI == 1) outh[o] = (_Float16)vv;
        else {
          float g = 0.5f * vv * (1.f + erff(vv * 0.7071067811865476f));
          outh[o] = (_Float16)g;
        }
      }
    }
}

// ----------------------------------------------------- fused sparse attn ---
// One block per (b, h, 16-query tile). 8 waves each own 128 keys.
__global__ __launch_bounds__(256) void k_attn(const _Float16* __restrict__ qg,
                                              const _Float16* __restrict__ kg,
                                              const _Float16* __restrict__ vg,
                                              _Float16* __restrict__ og) {
  __shared__ _Float16 Qs[16 * 64];       // 2 KB
  __shared__ _Float16 sc[16][1024];      // 32 KB: scores -> probs (f16)
  __shared__ float part[16][64];         // 4 KB: cross-wave AV reduction

  const int t = threadIdx.x, lane = t & 31, wid = t >> 5;
  const int hi = lane >> 4, l16 = lane & 15;
  int bid = blockIdx.x;
  int qt = bid & 63;
  int bh = bid >> 6;
  int h = bh & (Hn - 1);
  int b = bh >> 3;
  const size_t baseq = ((size_t)(b * Tsz + qt * 16) * Hn + h) * Dh;
  const size_t basek = ((size_t)b * Tsz * Hn + h) * Dh;  // + key*Hn*Dh

  for (int i = t; i < 16 * 64; i += 256) {
    int m = i >> 6, d = i & 63;
    Qs[m * 64 + d] = qg[baseq + (size_t)m * Hn * Dh + d];
    ((float*)part)[i] = 0.f;
  }
  __syncthreads();

  // ---- scores: 16 queries x 128 keys per wave, via WMMA (K = Dh = 64)
  const float scale = 0.125f;            // 1/sqrt(64)
  for (int tt = 0; tt < 8; ++tt) {
    int keyb = wid * 128 + tt * 16;
    v8f acc = {0.f, 0.f, 0.f, 0.f, 0.f, 0.f, 0.f, 0.f};
#pragma unroll
    for (int s = 0; s < 2; ++s) {
      v16h af;
      unsigned* au = (unsigned*)&af;
      const _Float16* qr = Qs + l16 * 64 + s * 32;
#pragma unroll
      for (int v = 0; v < 8; ++v)
        au[v] = *(const unsigned*)(qr + 2 * v + (v >= 4 ? 8 : 0) + hi * 8);
      v16h bfg;
      unsigned* bu = (unsigned*)&bfg;
      const _Float16* kr = kg + basek + (size_t)(keyb + l16) * Hn * Dh + s * 32;
#pragma unroll
      for (int v = 0; v < 8; ++v)
        bu[v] = *(const unsigned*)(kr + 2 * v + (v >= 4 ? 8 : 0) + hi * 8);
      acc = wmma_f16(af, bfg, acc);
    }
#pragma unroll
    for (int r = 0; r < 8; ++r)
      sc[r + hi * 8][keyb + l16] = (_Float16)(acc[r] * scale);
  }
  __syncthreads();

  // ---- per-row top-32 threshold + softmax (group of 16 lanes per row)
  {
    int row = t >> 4, j = t & 15;
    float cur = __builtin_inff(), rowmax = 0.f;
    for (int it = 0; it < TOPK; ++it) {
      float mx = -__builtin_inff();
      for (int c = j; c < Tsz; c += 16) {
        float s = (float)sc[row][c];
        if (s < cur && s > mx) mx = s;
      }
#pragma unroll
      for (int m = 8; m; m >>= 1) mx = fmaxf(mx, __shfl_xor(mx, m, 16));
      if (it == 0) rowmax = mx;
      cur = mx;
    }
    float th = cur;
    float sum = 0.f;
    for (int c = j; c < Tsz; c += 16) {
      float s = (float)sc[row][c];
      float pp = (s >= th) ? __expf(s - rowmax) : 0.f;
      sc[row][c] = (_Float16)pp;
      sum += pp;
    }
#pragma unroll
    for (int m = 8; m; m >>= 1) sum += __shfl_xor(sum, m, 16);
    float inv = 1.f / sum;
    for (int c = j; c < Tsz; c += 16)
      sc[row][c] = (_Float16)((float)sc[row][c] * inv);
  }
  __syncthreads();

  // ---- out = probs @ V  (each wave: its 128 keys; LDS-atomic reduce)
  {
    v8f acc[4];
    const v8f vz = {0.f, 0.f, 0.f, 0.f, 0.f, 0.f, 0.f, 0.f};
#pragma unroll
    for (int j = 0; j < 4; ++j) acc[j] = vz;
    int keyw = wid * 128;
    for (int kc = 0; kc < 4; ++kc) {
      int kb2 = keyw + kc * 32;
      v16h af;
      unsigned* au = (unsigned*)&af;
      const _Float16* pr = &sc[l16][kb2];
#pragma unroll
      for (int v = 0; v < 8; ++v)
        au[v] = *(const unsigned*)(pr + 2 * v + hi * 16);
#pragma unroll
      for (int j = 0; j < 4; ++j) {
        v16h bfg;
        const _Float16* vr =
            vg + basek + (size_t)(kb2 + hi * 16) * Hn * Dh + j * 16 + l16;
#pragma unroll
        for (int w = 0; w < 8; ++w) {
          bfg[2 * w]     = vr[(size_t)(2 * w) * Hn * Dh];
          bfg[2 * w + 1] = vr[(size_t)(2 * w + 1) * Hn * Dh];
        }
        acc[j] = wmma_f16(af, bfg, acc[j]);
      }
    }
#pragma unroll
    for (int j = 0; j < 4; ++j)
#pragma unroll
      for (int r = 0; r < 8; ++r)
        atomicAdd(&part[r + hi * 8][j * 16 + l16], acc[j][r]);
  }
  __syncthreads();

  for (int i = t; i < 16 * 64; i += 256) {
    int m = i >> 6, d = i & 63;
    og[baseq + (size_t)m * Hn * Dh + d] = (_Float16)part[m][d];
  }
}

// -------------------------------------------------- residual + LayerNorm ---
__global__ __launch_bounds__(256) void k_addln(float* __restrict__ hf,
                                               const float* __restrict__ ao,
                                               const float* __restrict__ g,
                                               const float* __restrict__ be,
                                               _Float16* __restrict__ hb) {
  __shared__ float rs0[8], rs1[8];
  int row = blockIdx.x, t = threadIdx.x;
  size_t base = (size_t)row * Dm;
  float x0 = hf[base + t] + ao[base + t];
  float x1 = hf[base + 256 + t] + ao[base + 256 + t];
  float s = x0 + x1, ss = x0 * x0 + x1 * x1;
#pragma unroll
  for (int m = 16; m; m >>= 1) {
    s += __shfl_xor(s, m, 32);
    ss += __shfl_xor(ss, m, 32);
  }
  if ((t & 31) == 0) { rs0[t >> 5] = s; rs1[t >> 5] = ss; }
  __syncthreads();
  s = 0.f; ss = 0.f;
#pragma unroll
  for (int i = 0; i < 8; ++i) { s += rs0[i]; ss += rs1[i]; }
  float mean = s * (1.f / Dm);
  float var = ss * (1.f / Dm) - mean * mean;
  float rstd = rsqrtf(var + 1e-5f);
  float y0 = (x0 - mean) * rstd * g[t] + be[t];
  float y1 = (x1 - mean) * rstd * g[256 + t] + be[256 + t];
  hf[base + t] = y0;       hf[base + 256 + t] = y1;
  hb[base + t] = (_Float16)y0; hb[base + 256 + t] = (_Float16)y1;
}

// ------------------------------------------------------------ dual heads ---
__global__ __launch_bounds__(256) void k_heads(const float* __restrict__ hf,
                                               const float* __restrict__ gm,
                                               const float* __restrict__ bm,
                                               const float* __restrict__ Wm,
                                               const float* __restrict__ cm,
                                               const float* __restrict__ gs,
                                               const float* __restrict__ bs,
                                               const float* __restrict__ Wsg,
                                               const float* __restrict__ cs,
                                               float* __restrict__ out) {
  __shared__ float r0[8], r1[8], r2[8], r3[8];
  int b = blockIdx.x, t = threadIdx.x;
  const float* x = hf + ((size_t)(b * Tsz + Tsz - 1)) * Dm;
  float x0 = x[t], x1 = x[t + 256];
  float s = x0 + x1, ss = x0 * x0 + x1 * x1;
#pragma unroll
  for (int m = 16; m; m >>= 1) {
    s += __shfl_xor(s, m, 32);
    ss += __shfl_xor(ss, m, 32);
  }
  if ((t & 31) == 0) { r0[t >> 5] = s; r1[t >> 5] = ss; }
  __syncthreads();
  s = 0.f; ss = 0.f;
#pragma unroll
  for (int i = 0; i < 8; ++i) { s += r0[i]; ss += r1[i]; }
  float mean = s * (1.f / Dm);
  float var = ss * (1.f / Dm) - mean * mean;
  float rstd = rsqrtf(var + 1e-5f);
  float n0 = (x0 - mean) * rstd, n1 = (x1 - mean) * rstd;
  float d1 = (n0 * gm[t] + bm[t]) * Wm[t] + (n1 * gm[t + 256] + bm[t + 256]) * Wm[t + 256];
  float d2 = (n0 * gs[t] + bs[t]) * Wsg[t] + (n1 * gs[t + 256] + bs[t + 256]) * Wsg[t + 256];
#pragma unroll
  for (int m = 16; m; m >>= 1) {
    d1 += __shfl_xor(d1, m, 32);
    d2 += __shfl_xor(d2, m, 32);
  }
  __syncthreads();
  if ((t & 31) == 0) { r2[t >> 5] = d1; r3[t >> 5] = d2; }
  __syncthreads();
  if (t == 0) {
    d1 = 0.f; d2 = 0.f;
#pragma unroll
    for (int i = 0; i < 8; ++i) { d1 += r2[i]; d2 += r3[i]; }
    const float Z99 = 2.326347874040841f;
    float mu = d1 + cm[0];
    float sig = 1.f / (1.f + __expf(-(d2 + cs[0])));
    float phi = __expf(-0.5f * Z99 * Z99) * 0.3989422804014327f;
    out[b] = mu;
    out[16 + b] = sig;
    out[32 + b] = mu + sig * Z99;
    out[48 + b] = mu + sig * phi * 100.f;   // /(1-CONF)
  }
}

// ------------------------------------------------------------------ host ---
extern "C" void kernel_launch(void* const* d_in, const int* in_sizes, int n_in,
                              void* d_out, int out_size, void* d_ws,
                              size_t ws_size, hipStream_t stream) {
  (void)in_sizes; (void)n_in; (void)out_size; (void)ws_size;
  const float* x    = (const float*)d_in[0];
  const float* W_in = (const float*)d_in[1];
  const float* b_in = (const float*)d_in[2];
  const float* Wq   = (const float*)d_in[3];
  const float* Wk   = (const float*)d_in[4];
  const float* Wv   = (const float*)d_in[5];
  const float* Wo   = (const float*)d_in[6];
  const float* bo   = (const float*)d_in[7];
  const float* W1   = (const float*)d_in[8];
  const float* b1   = (const float*)d_in[9];
  const float* W2   = (const float*)d_in[10];
  const float* b2   = (const float*)d_in[11];
  const float* g1   = (const float*)d_in[12];
  const float* be1  = (const float*)d_in[13];
  const float* g2   = (const float*)d_in[14];
  const float* be2  = (const float*)d_in[15];
  const float* gm   = (const float*)d_in[16];
  const float* bm   = (const float*)d_in[17];
  const float* Wm   = (const float*)d_in[18];
  const float* cm   = (const float*)d_in[19];
  const float* gs   = (const float*)d_in[20];
  const float* bs   = (const float*)d_in[21];
  const float* Wsg  = (const float*)d_in[22];
  const float* cs   = (const float*)d_in[23];

  char* p = (char*)d_ws;
  auto alloc = [&](size_t bytes) -> void* {
    void* r = (void*)p;
    p += (bytes + 255) & ~(size_t)255;
    return r;
  };
  float*    hf  = (float*)alloc((size_t)Ntok * Dm * 4);
  float*    ao  = (float*)alloc((size_t)Ntok * Dm * 4);
  _Float16* hb  = (_Float16*)alloc((size_t)Ntok * Dm * 2);
  _Float16* qb  = (_Float16*)alloc((size_t)Ntok * Dm * 2);
  _Float16* kb  = (_Float16*)alloc((size_t)Ntok * Dm * 2);
  _Float16* vb  = (_Float16*)alloc((size_t)Ntok * Dm * 2);
  _Float16* gb  = (_Float16*)alloc((size_t)Ntok * DffC * 2);
  _Float16* wqh = (_Float16*)alloc((size_t)Lc * Dm * Dm * 2);
  _Float16* wkh = (_Float16*)alloc((size_t)Lc * Dm * Dm * 2);
  _Float16* wvh = (_Float16*)alloc((size_t)Lc * Dm * Dm * 2);
  _Float16* woh = (_Float16*)alloc((size_t)Lc * Dm * Dm * 2);
  _Float16* w1h = (_Float16*)alloc((size_t)Lc * Dm * DffC * 2);
  _Float16* w2h = (_Float16*)alloc((size_t)Lc * DffC * Dm * 2);

  k_cvt<<<2048, 256, 0, stream>>>(Wq, wqh, Lc * Dm * Dm);
  k_cvt<<<2048, 256, 0, stream>>>(Wk, wkh, Lc * Dm * Dm);
  k_cvt<<<2048, 256, 0, stream>>>(Wv, wvh, Lc * Dm * Dm);
  k_cvt<<<2048, 256, 0, stream>>>(Wo, woh, Lc * Dm * Dm);
  k_cvt<<<4096, 256, 0, stream>>>(W1, w1h, Lc * Dm * DffC);
  k_cvt<<<4096, 256, 0, stream>>>(W2, w2h, Lc * DffC * Dm);

  k_inproj<<<(Ntok * Dm) / 256, 256, 0, stream>>>(x, W_in, b_in, hf, hb);

  dim3 gD(Dm / 128, Ntok / 128);      // (4, 128)
  dim3 gF(DffC / 128, Ntok / 128);    // (16, 128)

  for (int i = 0; i < Lc; ++i) {
    const _Float16* wq_i = wqh + (size_t)i * Dm * Dm;
    const _Float16* wk_i = wkh + (size_t)i * Dm * Dm;
    const _Float16* wv_i = wvh + (size_t)i * Dm * Dm;
    const _Float16* wo_i = woh + (size_t)i * Dm * Dm;
    const _Float16* w1_i = w1h + (size_t)i * Dm * DffC;
    const _Float16* w2_i = w2h + (size_t)i * DffC * Dm;

    k_gemm<1><<<gD, 256, 0, stream>>>(hb, wq_i, nullptr, nullptr, qb, Ntok, Dm, Dm);
    k_gemm<1><<<gD, 256, 0, stream>>>(hb, wk_i, nullptr, nullptr, kb, Ntok, Dm, Dm);
    k_gemm<1><<<gD, 256, 0, stream>>>(hb, wv_i, nullptr, nullptr, vb, Ntok, Dm, Dm);

    k_attn<<<Bsz * Hn * (Tsz / 16), 256, 0, stream>>>(qb, kb, vb, hb);

    k_gemm<0><<<gD, 256, 0, stream>>>(hb, wo_i, bo + (size_t)i * Dm, ao, nullptr,
                                      Ntok, Dm, Dm);
    k_addln<<<Ntok, 256, 0, stream>>>(hf, ao, g1 + (size_t)i * Dm,
                                      be1 + (size_t)i * Dm, hb);

    k_gemm<2><<<gF, 256, 0, stream>>>(hb, w1_i, b1 + (size_t)i * DffC, nullptr,
                                      gb, Ntok, Dm, DffC);
    k_gemm<0><<<gD, 256, 0, stream>>>(gb, w2_i, b2 + (size_t)i * Dm, ao, nullptr,
                                      Ntok, DffC, Dm);
    k_addln<<<Ntok, 256, 0, stream>>>(hf, ao, g2 + (size_t)i * Dm,
                                      be2 + (size_t)i * Dm, hb);
  }

  k_heads<<<Bsz, 256, 0, stream>>>(hf, gm, bm, Wm, cm, gs, bs, Wsg, cs,
                                   (float*)d_out);
}
